// LocallyEnhancedAttention_43980465111390
// MI455X (gfx1250) — matrix operand order repaired and checked by reading, not compiled, not run on hardware
//
#include <hip/hip_runtime.h>
#include <hip/hip_bf16.h>
#include <cstdint>
#include <math.h>

typedef __attribute__((ext_vector_type(16))) _Float16 v16h;
typedef __attribute__((ext_vector_type(8)))  _Float16 v8h;
typedef __attribute__((ext_vector_type(8)))  float    v8f;

#define BB    8
#define HHH   56
#define WWW   56
#define NNN   (HHH*WWW)   // 3136
#define DIMM  512
#define NHH   8
#define SRR   7
#define HDIMM 256
#define LDIMM 256
#define DHH   32
#define NQQ   64          // 8x8 pooled grid
#define EPSS  1e-5f

__device__ __forceinline__ float gelu_exact(float x) {
    return 0.5f * x * (1.0f + erff(x * 0.70710678118654752f));
}
__device__ __forceinline__ int iclampi(int v, int lo, int hi) {
    return v < lo ? lo : (v > hi ? hi : v);
}

// ---------------- prep kernels ----------------
__global__ void k_bn_prep(const float* g, const float* b, const float* m,
                          const float* v, float* scale, float* shift, int C) {
    int i = blockIdx.x * blockDim.x + threadIdx.x;
    if (i < C) {
        float s = g[i] * rsqrtf(v[i] + EPSS);
        scale[i] = s;
        shift[i] = b[i] - m[i] * s;
    }
}

__global__ void k_cvt_f16(const float* in, _Float16* out, int n) {
    int i = blockIdx.x * blockDim.x + threadIdx.x;
    if (i < n) out[i] = (_Float16)in[i];
}

// x (25088 x 512) -> first 256 cols as f16 (25088 x 256)
__global__ void k_extract_ahigh(const float* __restrict__ x, _Float16* __restrict__ a) {
    int row = blockIdx.x;
    int c = threadIdx.x;
    a[(long)row * HDIMM + c] = (_Float16)x[(long)row * DIMM + c];
}

// ---------------- generic batched WMMA GEMM ----------------
// C = A(MxK, f16, row stride lda) * Bw(NxK, f16, row stride ldb)^T
// optional: + bias[n]; then *bscale[n]+bshift[n]; then gelu; write f32 and/or f16.
template<int NW>
__global__ void k_gemm_wmma(const _Float16* __restrict__ A,
                            const _Float16* __restrict__ Bw,
                            float* __restrict__ C32, _Float16* __restrict__ C16,
                            const float* __restrict__ bias,
                            const float* __restrict__ bscale,
                            const float* __restrict__ bshift,
                            int M, int N, int K,
                            int lda, int ldb, int ldc32, int ldc16,
                            long sA, long sB, long sC32, long sC16,
                            int do_gelu) {
    const int lane = threadIdx.x & 31;
    const int wave = threadIdx.x >> 5;
    const int tiles_n = N / (16 * NW);
    const int tiles_m = M / 16;
    const int t = blockIdx.x * (blockDim.x >> 5) + wave;
    if (t >= tiles_m * tiles_n) return;
    const int tm = t / tiles_n, tn = t % tiles_n;
    const int batch = blockIdx.y;
    const _Float16* Ab = A + (long)batch * sA;
    const _Float16* Bb = Bw + (long)batch * sB;
    const int m0 = tm * 16, n0 = tn * 16 * NW;
    const int lm = lane & 15, lhi = lane >> 4;

    const v8f z8 = {0.f, 0.f, 0.f, 0.f, 0.f, 0.f, 0.f, 0.f};
    v8f acc[NW];
#pragma unroll
    for (int s = 0; s < NW; s++) acc[s] = z8;

    // A fragment rows: lane L -> M = L%16 ; halves 0..7: K=8*lhi+j ; halves 8..15: +16
    const _Float16* arow = Ab + (long)(m0 + lm) * lda + lhi * 8;
    for (int k0 = 0; k0 < K; k0 += 32) {
        v8h alo = *(const v8h*)(arow + k0);
        v8h ahi = *(const v8h*)(arow + k0 + 16);
        v16h afrag = __builtin_shufflevector(alo, ahi,
            0,1,2,3,4,5,6,7,8,9,10,11,12,13,14,15);
#pragma unroll
        for (int s = 0; s < NW; s++) {
            // B fragment: lane L -> N = L%16 ; halves 0..7: K=8*lhi+j ; halves 8..15: +16
            const _Float16* brow = Bb + (long)(n0 + s * 16 + lm) * ldb + k0 + lhi * 8;
            v8h blo = *(const v8h*)(brow);
            v8h bhi = *(const v8h*)(brow + 16);
            v16h bfrag = __builtin_shufflevector(blo, bhi,
                0,1,2,3,4,5,6,7,8,9,10,11,12,13,14,15);
            acc[s] = __builtin_amdgcn_wmma_f32_16x16x32_f16(
                false, afrag, false, bfrag, (short)0, acc[s], false, false);
        }
    }

#pragma unroll
    for (int s = 0; s < NW; s++) {
        const int n = n0 + s * 16 + lm;
        const float bi = bias ? bias[n] : 0.0f;
        const float sc = bscale ? bscale[n] : 1.0f;
        const float sh = bshift ? bshift[n] : 0.0f;
#pragma unroll
        for (int r = 0; r < 8; r++) {
            const int m = m0 + r + 8 * lhi;   // C/D layout: VGPR r, lanes>=16 -> M+8
            float val = acc[s][r] + bi;
            if (bscale) val = val * sc + sh;
            if (do_gelu) val = gelu_exact(val);
            if (C32) C32[(long)batch * sC32 + (long)m * ldc32 + n] = val;
            if (C16) C16[(long)batch * sC16 + (long)m * ldc16 + n] = (_Float16)val;
        }
    }
}

// ---------------- depthwise 3x3 conv + BN (NHWC f32 in) ----------------
__global__ void k_dwconv_bn(const float* __restrict__ in, const float* __restrict__ w,
                            const float* __restrict__ scale, const float* __restrict__ shift,
                            float* __restrict__ out32,
                            _Float16* __restrict__ out16, int ld16, int col_off,
                            int gelu_in) {
    int idx = blockIdx.x * blockDim.x + threadIdx.x;   // (b,n,c), c fastest
    int c = idx & 255;
    int p = idx >> 8;                                   // b*NNN + n
    int b = p / NNN;
    int n = p - b * NNN;
    int y = n / WWW, x = n - y * WWW;
    float acc = 0.f;
#pragma unroll
    for (int dy = -1; dy <= 1; dy++) {
        int yy = y + dy;
        if (yy < 0 || yy >= HHH) continue;
#pragma unroll
        for (int dx = -1; dx <= 1; dx++) {
            int xx = x + dx;
            if (xx < 0 || xx >= WWW) continue;
            float v = in[((long)(b * NNN + yy * WWW + xx)) * LDIMM + c];
            if (gelu_in) v = gelu_exact(v);
            acc += v * w[c * 9 + (dy + 1) * 3 + (dx + 1)];
        }
    }
    float r = acc * scale[c] + shift[c];
    if (out32) out32[(long)p * LDIMM + c] = r;
    if (out16) out16[(long)p * ld16 + col_off + c] = (_Float16)r;
}

// ---------------- 7x7 avg + max pool: NHWC f32 in -> (b, c, 64) ----------------
__global__ void k_pool(const float* __restrict__ in,
                       float* __restrict__ avg, float* __restrict__ mx) {
    int idx = blockIdx.x * blockDim.x + threadIdx.x;  // (b, q, c), c fastest
    int c = idx & 255;
    int q = (idx >> 8) & 63;
    int b = idx >> 14;
    int qy = q >> 3, qx = q & 7;
    float s = 0.f, m = -3.4e38f;
    for (int i = 0; i < SRR; i++)
        for (int j = 0; j < SRR; j++) {
            float v = in[((long)(b * NNN + (qy * SRR + i) * WWW + qx * SRR + j)) * LDIMM + c];
            s += v;
            m = fmaxf(m, v);
        }
    long o = ((long)(b * LDIMM + c)) * NQQ + q;
    avg[o] = s * (1.0f / 49.0f);
    mx[o] = m;
}

// ---------------- v (B,N,C) f32 -> vT (B*C, N) f16 ----------------
__global__ void k_vT(const float* __restrict__ v32, _Float16* __restrict__ vT) {
    long idx = (long)blockIdx.x * blockDim.x + threadIdx.x;  // (b, c, n), n fastest
    int n = (int)(idx % NNN);
    long r = idx / NNN;
    int c = (int)(r % LDIMM);
    int b = (int)(r / LDIMM);
    vT[idx] = (_Float16)v32[((long)(b * NNN + n)) * LDIMM + c];
}

// ---------------- attn = q@k^T on pooled grids ----------------
__global__ void k_attn_qk(const float* __restrict__ q1, const float* __restrict__ q2,
                          const float* __restrict__ k1, const float* __restrict__ k2,
                          float* __restrict__ attn) {
    int idx = blockIdx.x * blockDim.x + threadIdx.x;  // ((b*8+h)*64+qi)*64+kj
    int kj = idx & 63;
    int qi = (idx >> 6) & 63;
    int h  = (idx >> 12) & 7;
    int b  = idx >> 15;
    float s = 0.f;
    int cbase = (b * LDIMM + h * DHH) * NQQ;
    for (int d = 0; d < DHH; d++) {
        int o = cbase + d * NQQ;
        s += q1[o + qi] * k1[o + kj] + q2[o + qi] * k2[o + kj];
    }
    attn[idx] = s;
}

// ---------------- bilinear 8x8 -> 56x56, *SCALE, softmax over 3136, f16 out ----------------
__global__ void k_softmax_resize(const float* __restrict__ attn_s,
                                 _Float16* __restrict__ out) {
    __shared__ float tile[NQQ];
    __shared__ float row[NNN];
    __shared__ float red[256];
    const int rid = blockIdx.x;      // (b*8+h)*64 + qi
    const int tid = threadIdx.x;
    if (tid < NQQ) tile[tid] = attn_s[(long)rid * NQQ + tid];
    __syncthreads();
    const float SCALE = 0.044194173824159216f;   // 512^-0.5
    float lmax = -3.4e38f;
    for (int n = tid; n < NNN; n += 256) {
        int y = n / WWW, x = n - y * WWW;
        float fy = (y + 0.5f) * (1.0f / 7.0f) - 0.5f;
        float fx = (x + 0.5f) * (1.0f / 7.0f) - 0.5f;
        float fyf = floorf(fy), fxf = floorf(fx);
        int y0 = (int)fyf, x0 = (int)fxf;
        float wy = fy - fyf, wx = fx - fxf;
        int y0c = iclampi(y0, 0, 7), y1c = iclampi(y0 + 1, 0, 7);
        int x0c = iclampi(x0, 0, 7), x1c = iclampi(x0 + 1, 0, 7);
        float v00 = tile[y0c * 8 + x0c], v01 = tile[y0c * 8 + x1c];
        float v10 = tile[y1c * 8 + x0c], v11 = tile[y1c * 8 + x1c];
        float v = (v00 * (1.f - wx) + v01 * wx) * (1.f - wy) +
                  (v10 * (1.f - wx) + v11 * wx) * wy;
        v *= SCALE;
        row[n] = v;
        lmax = fmaxf(lmax, v);
    }
    red[tid] = lmax;
    __syncthreads();
    for (int s = 128; s > 0; s >>= 1) {
        if (tid < s) red[tid] = fmaxf(red[tid], red[tid + s]);
        __syncthreads();
    }
    const float mx = red[0];
    __syncthreads();
    float lsum = 0.f;
    for (int n = tid; n < NNN; n += 256) {
        float e = __expf(row[n] - mx);
        row[n] = e;
        lsum += e;
    }
    red[tid] = lsum;
    __syncthreads();
    for (int s = 128; s > 0; s >>= 1) {
        if (tid < s) red[tid] += red[tid + s];
        __syncthreads();
    }
    const float inv = 1.0f / red[0];
    __syncthreads();
    for (int n = tid; n < NNN; n += 256)
        out[(long)rid * NNN + n] = (_Float16)(row[n] * inv);
}

// ---------------- bilinear upsample attn-out (B,8,64,32) -> NHWC f32 (B,N,256) ----------------
__global__ void k_upsample(const float* __restrict__ ao, float* __restrict__ up) {
    int idx = blockIdx.x * blockDim.x + threadIdx.x;  // (b,n,c), c fastest
    int c = idx & 255;
    int p = idx >> 8;
    int b = p / NNN;
    int n = p - b * NNN;
    int y = n / WWW, x = n - y * WWW;
    int h = c >> 5, d = c & 31;
    float fy = (y + 0.5f) * (1.0f / 7.0f) - 0.5f;
    float fx = (x + 0.5f) * (1.0f / 7.0f) - 0.5f;
    float fyf = floorf(fy), fxf = floorf(fx);
    int y0 = (int)fyf, x0 = (int)fxf;
    float wy = fy - fyf, wx = fx - fxf;
    int y0c = iclampi(y0, 0, 7), y1c = iclampi(y0 + 1, 0, 7);
    int x0c = iclampi(x0, 0, 7), x1c = iclampi(x0 + 1, 0, 7);
    long base = ((long)(b * NHH + h)) * NQQ;        // ao: ((b*8+h)*64+q)*32+d
    float v00 = ao[(base + y0c * 8 + x0c) * DHH + d];
    float v01 = ao[(base + y0c * 8 + x1c) * DHH + d];
    float v10 = ao[(base + y1c * 8 + x0c) * DHH + d];
    float v11 = ao[(base + y1c * 8 + x1c) * DHH + d];
    float v = (v00 * (1.f - wx) + v01 * wx) * (1.f - wy) +
              (v10 * (1.f - wx) + v11 * wx) * wy;
    up[(long)p * LDIMM + c] = v;
}

extern "C" void kernel_launch(void* const* d_in, const int* in_sizes, int n_in,
                              void* d_out, int out_size, void* d_ws, size_t ws_size,
                              hipStream_t stream) {
    (void)in_sizes; (void)n_in; (void)out_size; (void)ws_size;
    const float* x        = (const float*)d_in[0];
    const float* proj_v_w = (const float*)d_in[1];
    const float* conv1_w  = (const float*)d_in[2];
    const float* conv2_w  = (const float*)d_in[3];
    const float* conv3_w  = (const float*)d_in[4];
    const float* conv3_b  = (const float*)d_in[5];
    const float* v_conv_w = (const float*)d_in[6];
    const float* proj_w   = (const float*)d_in[7];
    const float* proj_b   = (const float*)d_in[8];
    const float* bn1_g = (const float*)d_in[9];
    const float* bn1_b = (const float*)d_in[10];
    const float* bn1_m = (const float*)d_in[11];
    const float* bn1_v = (const float*)d_in[12];
    const float* bn2_g = (const float*)d_in[13];
    const float* bn2_b = (const float*)d_in[14];
    const float* bn2_m = (const float*)d_in[15];
    const float* bn2_v = (const float*)d_in[16];
    const float* bn3_g = (const float*)d_in[17];
    const float* bn3_b = (const float*)d_in[18];
    const float* bn3_m = (const float*)d_in[19];
    const float* bn3_v = (const float*)d_in[20];
    const float* vbn_g = (const float*)d_in[21];
    const float* vbn_b = (const float*)d_in[22];
    const float* vbn_m = (const float*)d_in[23];
    const float* vbn_v = (const float*)d_in[24];

    // bump allocator in workspace (256B aligned)
    uint8_t* base = (uint8_t*)d_ws;
    size_t off = 0;
    auto alloc = [&](size_t bytes) -> void* {
        void* p = base + off;
        off = (off + bytes + 255) & ~(size_t)255;
        return p;
    };

    const size_t M = (size_t)BB * NNN;   // 25088

    _Float16* Wv16 = (_Float16*)alloc((size_t)LDIMM * HDIMM * 2);
    _Float16* W316 = (_Float16*)alloc((size_t)HDIMM * LDIMM * 2);
    _Float16* Wp16 = (_Float16*)alloc((size_t)DIMM * DIMM * 2);
    float* bn1s = (float*)alloc(LDIMM * 4); float* bn1h = (float*)alloc(LDIMM * 4);
    float* bn2s = (float*)alloc(LDIMM * 4); float* bn2h = (float*)alloc(LDIMM * 4);
    float* bn3s = (float*)alloc(HDIMM * 4); float* bn3h = (float*)alloc(HDIMM * 4);
    float* vbns = (float*)alloc(LDIMM * 4); float* vbnh = (float*)alloc(LDIMM * 4);
    _Float16* Ahigh = (_Float16*)alloc(M * HDIMM * 2);   // later reused as xh2_f16
    float* v32  = (float*)alloc(M * LDIMM * 4);          // later reused as upsampled
    float* xh1  = (float*)alloc(M * LDIMM * 4);          // later reused as attn_f16
    float* xh2  = (float*)alloc(M * LDIMM * 4);
    float* q1 = (float*)alloc((size_t)BB * LDIMM * NQQ * 4);
    float* q2 = (float*)alloc((size_t)BB * LDIMM * NQQ * 4);
    float* k1 = (float*)alloc((size_t)BB * LDIMM * NQQ * 4);
    float* k2 = (float*)alloc((size_t)BB * LDIMM * NQQ * 4);
    float* attn_s = (float*)alloc((size_t)BB * NHH * NQQ * NQQ * 4);
    _Float16* vT16 = (_Float16*)alloc((size_t)BB * LDIMM * NNN * 2);
    float* aout = (float*)alloc((size_t)BB * NHH * NQQ * DHH * 4);
    _Float16* Acat = (_Float16*)alloc(M * DIMM * 2);

    _Float16* xh2_16 = Ahigh;          // alias: Ahigh dead after gemm_v
    float* up32 = v32;                 // alias: v32 dead after vT + dwconv1
    _Float16* attn16 = (_Float16*)xh1; // alias: xh1 dead after dwconv2

    // 1) BN folding + weight conversion + A extraction
    k_bn_prep<<<1, 256, 0, stream>>>(bn1_g, bn1_b, bn1_m, bn1_v, bn1s, bn1h, LDIMM);
    k_bn_prep<<<1, 256, 0, stream>>>(bn2_g, bn2_b, bn2_m, bn2_v, bn2s, bn2h, LDIMM);
    k_bn_prep<<<1, 256, 0, stream>>>(bn3_g, bn3_b, bn3_m, bn3_v, bn3s, bn3h, HDIMM);
    k_bn_prep<<<1, 256, 0, stream>>>(vbn_g, vbn_b, vbn_m, vbn_v, vbns, vbnh, LDIMM);
    k_cvt_f16<<<256, 256, 0, stream>>>(proj_v_w, Wv16, LDIMM * HDIMM);
    k_cvt_f16<<<256, 256, 0, stream>>>(conv3_w, W316, HDIMM * LDIMM);
    k_cvt_f16<<<1024, 256, 0, stream>>>(proj_w, Wp16, DIMM * DIMM);
    k_extract_ahigh<<<(int)M, 256, 0, stream>>>(x, Ahigh);

    // 2) v = x_high @ Wv^T  (25088 x 256, K=256)  [WMMA]
    k_gemm_wmma<4><<<dim3(784, 1), 256, 0, stream>>>(
        Ahigh, Wv16, v32, nullptr, nullptr, nullptr, nullptr,
        (int)M, 256, 256, 256, 256, 256, 0, 0, 0, 0, 0, 0);

    // 3) vT (f16, per-head K-contiguous) for attention-out GEMM
    k_vT<<<25088, 256, 0, stream>>>(v32, vT16);

    // 4) dwconv1 + bn1 -> xh1
    k_dwconv_bn<<<25088, 256, 0, stream>>>(v32, conv1_w, bn1s, bn1h, xh1,
                                           nullptr, 0, 0, 0);
    // 5) q pools
    k_pool<<<512, 256, 0, stream>>>(xh1, q1, q2);

    // 6) gelu -> dwconv2 + bn2 -> xh2 (f32 + f16)
    k_dwconv_bn<<<25088, 256, 0, stream>>>(xh1, conv2_w, bn2s, bn2h, xh2,
                                           xh2_16, LDIMM, 0, 1);
    // 7) k pools
    k_pool<<<512, 256, 0, stream>>>(xh2, k1, k2);

    // 8) conv3 1x1 + bias + bn3 + gelu -> x_high (f16 into concat cols 0..255) [WMMA]
    k_gemm_wmma<4><<<dim3(784, 1), 256, 0, stream>>>(
        xh2_16, W316, nullptr, Acat, conv3_b, bn3s, bn3h,
        (int)M, 256, 256, 256, 256, 0, DIMM, 0, 0, 0, 0, 1);

    // 9) attention logits on pooled grids
    k_attn_qk<<<1024, 256, 0, stream>>>(q1, q2, k1, k2, attn_s);

    // 10) bilinear-resize + scale + softmax -> f16 attention matrix
    k_softmax_resize<<<4096, 256, 0, stream>>>(attn_s, attn16);

    // 11) out = attn @ vh : 64 batches of (64 x 32, K=3136) [WMMA]
    k_gemm_wmma<2><<<dim3(1, BB * NHH), 256, 0, stream>>>(
        attn16, vT16, aout, nullptr, nullptr, nullptr, nullptr,
        NQQ, DHH, NNN, NNN, NNN, DHH, 0,
        (long)NQQ * NNN, (long)DHH * NNN, (long)NQQ * DHH, 0, 0);

    // 12) bilinear upsample 8x8 -> 56x56 (NHWC f32)
    k_upsample<<<25088, 256, 0, stream>>>(aout, up32);

    // 13) dwconv(v_conv) + vbn -> x_low (f16 into concat cols 256..511)
    k_dwconv_bn<<<25088, 256, 0, stream>>>(up32, v_conv_w, vbns, vbnh, nullptr,
                                           Acat, DIMM, HDIMM, 0);

    // 14) y = [x_high, x_low] @ proj_w^T + proj_b  (25088 x 512, K=512) [WMMA]
    k_gemm_wmma<4><<<dim3(1568, 1), 256, 0, stream>>>(
        Acat, Wp16, (float*)d_out, nullptr, proj_b, nullptr, nullptr,
        (int)M, DIMM, DIMM, DIMM, DIMM, DIMM, 0, 0, 0, 0, 0, 0);
}
